// GCN_4045859193302
// MI455X (gfx1250) — matrix operand order repaired and checked by reading, not compiled, not run on hardware
//
#include <hip/hip_runtime.h>

typedef __attribute__((ext_vector_type(16))) __bf16 v16bf;
typedef __attribute__((ext_vector_type(8)))  float  v8f;

#define NTW 4  // N-tiles (16-wide) per wave in the WMMA GEMM

#define WMMA_BF16(A, B, Cacc)                                               \
  __builtin_amdgcn_wmma_f32_16x16x32_bf16(false, (A), false, (B), (short)0, \
                                          (Cacc), false, false)

// ---------------------------------------------------------------- utilities
__global__ __launch_bounds__(256) void fill_f32(float* __restrict__ p, float v,
                                                long long n) {
  long long i = (long long)blockIdx.x * blockDim.x + threadIdx.x;
  if (i < n) p[i] = v;
}

__global__ __launch_bounds__(256) void deg_add(const long long* __restrict__ dst,
                                               float* __restrict__ deg, int nE) {
  int e = blockIdx.x * blockDim.x + threadIdx.x;
  if (e < nE) atomicAdd(&deg[dst[e]], 1.0f);
}

__global__ __launch_bounds__(256) void rsqrt_ip(float* __restrict__ p, int n) {
  int i = blockIdx.x * blockDim.x + threadIdx.x;
  if (i < n) {
    float d = p[i];
    p[i] = d > 0.0f ? rsqrtf(d) : 0.0f;
  }
}

// ------------------------------------------------ fragment packing (f32->bf16)
// A [M,K] row-major f32 -> per-(mTile,kTile,lane) 16 contiguous bf16 in the
// WMMA A-fragment layout (16-bit A 16x32). Fragment idx = (tm*kt + tk)*32+lane.
__global__ __launch_bounds__(256) void pack_a(const float* __restrict__ A,
                                              __bf16* __restrict__ P,
                                              int M, int K) {
  int idx = blockIdx.x * blockDim.x + threadIdx.x;
  int mt = (M + 15) >> 4, kt = K >> 5;
  if (idx >= mt * kt * 32) return;
  int lane = idx & 31;
  int tile = idx >> 5;
  int tk = tile % kt, tm = tile / kt;
  int half = lane >> 4, l = lane & 15;
  int row = tm * 16 + l;
  __bf16* o = P + (size_t)idx * 16;
  if (row >= M) {
#pragma unroll
    for (int j = 0; j < 16; ++j) o[j] = (__bf16)0.0f;
    return;
  }
  const float* arow = A + (size_t)row * K + tk * 32;
#pragma unroll
  for (int v = 0; v < 8; ++v) {
    // lanes 0-15: K = 0-15 ; lanes 16-31: K = 16-31 (pairs)
    int kb = 16 * half + 2 * v;
    o[2 * v]     = (__bf16)arow[kb];
    o[2 * v + 1] = (__bf16)arow[kb + 1];
  }
}

// B [K,N] row-major f32 -> per-(kTile,nTile,lane) 16 contiguous bf16 in the
// contiguous-k B-fragment layout matching pack_a (32x16: lanes0-15 K=0-15, lanes16-31 K=16-31).
// Fragment idx = (tk*nt + tn)*32 + lane.
__global__ __launch_bounds__(256) void pack_b(const float* __restrict__ B,
                                              __bf16* __restrict__ P,
                                              int K, int N) {
  int idx = blockIdx.x * blockDim.x + threadIdx.x;
  int kt = K >> 5, nt = N >> 4;
  if (idx >= kt * nt * 32) return;
  int lane = idx & 31;
  int tile = idx >> 5;
  int tn = tile % nt, tk = tile / nt;
  int half = lane >> 4, l = lane & 15;
  const float* bp = B + (size_t)(tk * 32) * N + tn * 16 + l;
  __bf16* o = P + (size_t)idx * 16;
#pragma unroll
  for (int v = 0; v < 8; ++v) {
    int kk = 16 * half + 2 * v;
    o[2 * v]     = (__bf16)bp[(size_t)kk * N];
    o[2 * v + 1] = (__bf16)bp[(size_t)(kk + 1) * N];
  }
}

// ------------------------------------------- WMMA GEMM on packed bf16 fragments
// One wave -> one 16-row M tile x NTW 16-col N tiles. Software-pipelined:
// next K-step's fragments load while current step's 4 WMMAs execute.
__global__ __launch_bounds__(128) void gemm_wmma_packed(
    const __bf16* __restrict__ Ap, const __bf16* __restrict__ Bp,
    float* __restrict__ C, int M, int K, int N) {
  unsigned wave = blockIdx.x * (blockDim.x >> 5) + (threadIdx.x >> 5);
  unsigned kt = (unsigned)K >> 5, nt = (unsigned)N >> 4;
  unsigned ng = nt / NTW;
  unsigned tm = wave / ng;
  unsigned tn0 = (wave % ng) * NTW;
  unsigned mt = ((unsigned)M + 15) >> 4;
  if (tm >= mt) return;  // wave-uniform: EXEC all-ones for WMMA

  unsigned lane = threadIdx.x & 31;
  // byte pointers; one fragment = 32 bytes; one tile = 32 fragments = 1024 B
  const char* ap = (const char*)Ap + ((size_t)tm * kt * 32 + lane) * 32;
  const char* bp = (const char*)Bp + ((size_t)tn0 * 32 + lane) * 32;
  const size_t astep = 1024;                   // one K-tile of A
  const size_t bstep = (size_t)nt * 1024;      // one K-tile row of B tiles

  v8f acc0 = (v8f){}, acc1 = (v8f){}, acc2 = (v8f){}, acc3 = (v8f){};

  v16bf a  = *(const v16bf*)ap;
  v16bf b0 = *(const v16bf*)(bp);
  v16bf b1 = *(const v16bf*)(bp + 1024);
  v16bf b2 = *(const v16bf*)(bp + 2048);
  v16bf b3 = *(const v16bf*)(bp + 3072);

  for (unsigned k = 1; k < kt; ++k) {
    ap += astep;
    bp += bstep;
    v16bf an = *(const v16bf*)ap;
    v16bf c0 = *(const v16bf*)(bp);
    v16bf c1 = *(const v16bf*)(bp + 1024);
    v16bf c2 = *(const v16bf*)(bp + 2048);
    v16bf c3 = *(const v16bf*)(bp + 3072);
    acc0 = WMMA_BF16(a, b0, acc0);
    acc1 = WMMA_BF16(a, b1, acc1);
    acc2 = WMMA_BF16(a, b2, acc2);
    acc3 = WMMA_BF16(a, b3, acc3);
    a = an; b0 = c0; b1 = c1; b2 = c2; b3 = c3;
  }
  acc0 = WMMA_BF16(a, b0, acc0);
  acc1 = WMMA_BF16(a, b1, acc1);
  acc2 = WMMA_BF16(a, b2, acc2);
  acc3 = WMMA_BF16(a, b3, acc3);

  unsigned half = lane >> 4, l = lane & 15;
  v8f accs[NTW] = {acc0, acc1, acc2, acc3};
#pragma unroll
  for (int j = 0; j < NTW; ++j) {
#pragma unroll
    for (int r = 0; r < 8; ++r) {
      unsigned row = tm * 16 + r + 8 * half;   // C: VGPR r -> M=r / r+8
      unsigned col = (tn0 + j) * 16 + l;
      C[(size_t)row * N + col] = accs[j][r];
    }
  }
}

// -------------------------------------------------------------- edge scatter
__global__ __launch_bounds__(256) void scatter_edges(
    const long long* __restrict__ src, const long long* __restrict__ dst,
    const float* __restrict__ dinv, const float* __restrict__ H,
    float* __restrict__ agg, int nE, int F) {
  int epb = blockDim.x >> 5;
  int e = blockIdx.x * epb + (threadIdx.x >> 5);
  if (e >= nE) return;
  int lane = threadIdx.x & 31;
  long long s = src[e], d = dst[e];
  float norm = dinv[s] * dinv[d];
  const float* hs = H + (size_t)s * F;
  float* ad = agg + (size_t)d * F;
  for (int f = lane; f < F; f += 32) atomicAdd(&ad[f], hs[f] * norm);
}

// --------------------------------------- self-loop + bias (+optional ReLU)
__global__ __launch_bounds__(256) void finish_node(
    const float* __restrict__ H, const float* __restrict__ dinv,
    const float* __restrict__ bias, float* __restrict__ agg,
    long long total, int F, int doRelu) {
  long long i = (long long)blockIdx.x * blockDim.x + threadIdx.x;
  if (i >= total) return;
  long long node = i / F;
  int f = (int)(i % F);
  float di = dinv[node];
  float v = agg[i] + H[i] * di * di + bias[f];
  agg[i] = doRelu ? fmaxf(v, 0.0f) : v;
}

// ------------------------------------------------------------------ pooling
__global__ __launch_bounds__(256) void pool_count(
    const long long* __restrict__ batch, float* __restrict__ cnt, int n) {
  int i = blockIdx.x * blockDim.x + threadIdx.x;
  if (i < n) atomicAdd(&cnt[batch[i]], 1.0f);
}

__global__ __launch_bounds__(256) void pool_sum(
    const long long* __restrict__ batch, const float* __restrict__ h,
    float* __restrict__ pooled, long long total, int F) {
  long long i = (long long)blockIdx.x * blockDim.x + threadIdx.x;
  if (i >= total) return;
  long long node = i / F;
  int f = (int)(i % F);
  atomicAdd(&pooled[(size_t)batch[node] * F + f], h[i]);
}

__global__ __launch_bounds__(256) void final_lin(
    const float* __restrict__ pooled, const float* __restrict__ cnt,
    const float* __restrict__ Wl, const float* __restrict__ bl,
    float* __restrict__ out, int G, int F, int Cc) {
  int idx = blockIdx.x * blockDim.x + threadIdx.x;
  if (idx >= G * Cc) return;
  int g = idx / Cc, c = idx % Cc;
  const float* pg = pooled + (size_t)g * F;
  float s = 0.0f;
  for (int k = 0; k < F; ++k) s += pg[k] * Wl[(size_t)k * Cc + c];
  out[idx] = s / fmaxf(cnt[g], 1.0f) + bl[c];
}

// -------------------------------------------------------------------- driver
extern "C" void kernel_launch(void* const* d_in, const int* in_sizes, int n_in,
                              void* d_out, int out_size, void* d_ws,
                              size_t ws_size, hipStream_t stream) {
  const float* x        = (const float*)d_in[0];
  const long long* ei   = (const long long*)d_in[1];
  const long long* batch= (const long long*)d_in[2];
  const float* W1 = (const float*)d_in[3];
  const float* b1 = (const float*)d_in[4];
  const float* W2 = (const float*)d_in[5];
  const float* b2 = (const float*)d_in[6];
  const float* Wl = (const float*)d_in[7];
  const float* bl = (const float*)d_in[8];
  float* out = (float*)d_out;

  const int Nn   = in_sizes[2];            // 50000 nodes
  const int E    = in_sizes[1] / 2;        // 800000 edges
  const int INF  = in_sizes[0] / Nn;       // 128
  const int HID  = in_sizes[4];            // 256
  const int NCLS = in_sizes[8];            // 10
  const int G    = out_size / NCLS;        // 256 graphs

  const long long* src = ei;
  const long long* dst = ei + E;

  auto al = [](size_t o) { return (o + 255) & ~(size_t)255; };
  char* ws = (char*)d_ws;
  size_t o = 0;
  float*  Hbuf   = (float*)(ws + o);  o = al(o + (size_t)Nn * HID * 4);
  float*  Agg    = (float*)(ws + o);  o = al(o + (size_t)Nn * HID * 4);
  __bf16* Apack  = (__bf16*)(ws + o); o = al(o + (size_t)Nn * HID * 2);
  __bf16* Bpack  = (__bf16*)(ws + o); o = al(o + (size_t)HID * HID * 2);
  float*  Dinv   = (float*)(ws + o);  o = al(o + (size_t)Nn * 4);
  float*  Pooled = (float*)(ws + o);  o = al(o + (size_t)G * HID * 4);
  float*  Cnt    = (float*)(ws + o);  o = al(o + (size_t)G * 4);
  (void)ws_size; (void)n_in;

  // degrees (self-loop contributes 1) -> dinv = rsqrt(deg)
  fill_f32<<<(Nn + 255) / 256, 256, 0, stream>>>(Dinv, 1.0f, Nn);
  deg_add<<<(E + 255) / 256, 256, 0, stream>>>(dst, Dinv, E);
  rsqrt_ip<<<(Nn + 255) / 256, 256, 0, stream>>>(Dinv, Nn);

  long long totNF = (long long)Nn * HID;
  unsigned nfBlocks = (unsigned)((totNF + 255) / 256);
  int mt = (Nn + 15) / 16, nt = HID / 16;
  unsigned gemmBlocks = (unsigned)(((long long)mt * (nt / NTW) + 3) / 4);

  // -------- layer 1: H = x @ W1 (WMMA) ; agg = norm-aggregate + b1 ; ReLU
  {
    int ktot1 = mt * (INF / 32) * 32;
    pack_a<<<(ktot1 + 255) / 256, 256, 0, stream>>>(x, Apack, Nn, INF);
    int btot1 = (INF / 32) * nt * 32;
    pack_b<<<(btot1 + 255) / 256, 256, 0, stream>>>(W1, Bpack, INF, HID);
    gemm_wmma_packed<<<gemmBlocks, 128, 0, stream>>>(Apack, Bpack, Hbuf, Nn,
                                                     INF, HID);
  }
  fill_f32<<<nfBlocks, 256, 0, stream>>>(Agg, 0.0f, totNF);
  scatter_edges<<<(E + 7) / 8, 256, 0, stream>>>(src, dst, Dinv, Hbuf, Agg, E, HID);
  finish_node<<<nfBlocks, 256, 0, stream>>>(Hbuf, Dinv, b1, Agg, totNF, HID, 1);

  // -------- layer 2: H = h @ W2 (WMMA) ; agg = norm-aggregate + b2
  {
    int ktot2 = mt * (HID / 32) * 32;
    pack_a<<<(ktot2 + 255) / 256, 256, 0, stream>>>(Agg, Apack, Nn, HID);
    int btot2 = (HID / 32) * nt * 32;
    pack_b<<<(btot2 + 255) / 256, 256, 0, stream>>>(W2, Bpack, HID, HID);
    gemm_wmma_packed<<<gemmBlocks, 128, 0, stream>>>(Apack, Bpack, Hbuf, Nn,
                                                     HID, HID);
  }
  fill_f32<<<nfBlocks, 256, 0, stream>>>(Agg, 0.0f, totNF);
  scatter_edges<<<(E + 7) / 8, 256, 0, stream>>>(src, dst, Dinv, Hbuf, Agg, E, HID);
  finish_node<<<nfBlocks, 256, 0, stream>>>(Hbuf, Dinv, b2, Agg, totNF, HID, 0);

  // -------- mean pool + linear head
  fill_f32<<<(unsigned)((G * HID + 255) / 256), 256, 0, stream>>>(
      Pooled, 0.0f, (long long)G * HID);
  fill_f32<<<(G + 255) / 256, 256, 0, stream>>>(Cnt, 0.0f, G);
  pool_count<<<(Nn + 255) / 256, 256, 0, stream>>>(batch, Cnt, Nn);
  pool_sum<<<nfBlocks, 256, 0, stream>>>(batch, Agg, Pooled, totNF, HID);
  final_lin<<<(G * NCLS + 255) / 256, 256, 0, stream>>>(Pooled, Cnt, Wl, bl, out,
                                                        G, HID, NCLS);
}
